// ViTClassifier_4587025072626
// MI455X (gfx1250) — matrix operand order repaired and checked by reading, not compiled, not run on hardware
//
#include <hip/hip_runtime.h>
#include <hip/hip_bf16.h>
#include <math.h>

// ---------------- problem constants ----------------
#define P_      16
#define D_      768
#define H_      12
#define L_      12
#define F_      3072
#define NC_     1000
#define B_      32
#define IMG_    224
#define GRID_   14
#define NPATCH_ 196            // 14*14
#define S_      197            // NPATCH_ + 1
#define DK_     64             // D_/H_
#define ROWS_   (B_ * S_)      // 6304
#define PROWS_  (B_ * NPATCH_) // 6272

typedef __attribute__((ext_vector_type(16))) __bf16 bf16x16;
typedef __attribute__((ext_vector_type(8)))  float  f32x8;
typedef __attribute__((address_space(3))) unsigned short lds_us_t;

union FragA { bf16x16 v; unsigned short us[16]; uint4 q[2]; };
union FragC { f32x8  v; float f[8]; };

// native bf16 convert (v_cvt on gfx1250), RNE
__device__ __forceinline__ unsigned short f2bf(float f) {
    union { __bf16 b; unsigned short u; } cv;
    cv.b = (__bf16)f;
    return cv.u;
}

// raw LDS byte offset for async-load VDST operand
__device__ __forceinline__ unsigned lds_off(const unsigned short* p) {
    return (unsigned)(unsigned long long)(lds_us_t*)p;
}

// async global->LDS 64B copy (4 x b128), CDNA5 GVS mode, tracked by ASYNCcnt
__device__ __forceinline__ void async_copy64B(const unsigned short* gbase,
                                              unsigned goff_bytes,
                                              unsigned ldsaddr) {
    asm volatile("global_load_async_to_lds_b128 %0, %1, %2 offset:0"
                 :: "v"(ldsaddr), "v"(goff_bytes), "s"(gbase) : "memory");
    asm volatile("global_load_async_to_lds_b128 %0, %1, %2 offset:16"
                 :: "v"(ldsaddr), "v"(goff_bytes), "s"(gbase) : "memory");
    asm volatile("global_load_async_to_lds_b128 %0, %1, %2 offset:32"
                 :: "v"(ldsaddr), "v"(goff_bytes), "s"(gbase) : "memory");
    asm volatile("global_load_async_to_lds_b128 %0, %1, %2 offset:48"
                 :: "v"(ldsaddr), "v"(goff_bytes), "s"(gbase) : "memory");
}
__device__ __forceinline__ void wait_async0() {
    asm volatile("s_wait_asynccnt 0" ::: "memory");
}

// =====================================================================
// bf16 WMMA GEMM:  out[M,N] = epi( A[M,K] @ Bt[N,K]^T + bias )
// A, Bt bf16; 64x64 block tile, BK=64, 128 thr = 4 waves (each 32x32).
// Double-buffered LDS staging via global_load_async_to_lds_b128:
//   issue tile t+1 into buf^1, compute tile t from buf, wait+barrier, swap.
// EPI: 0 = bias, 1 = bias + exact GELU, 2 = bias + residual add
// OUTBF: 1 -> write bf16, 0 -> write f32
// =====================================================================
template<int EPI, int OUTBF>
__global__ __launch_bounds__(128) void gemm_bf16_kernel(
    const unsigned short* __restrict__ A,  int lda,   // bf16 [M][lda]
    const unsigned short* __restrict__ Bt, int ldb,   // bf16 [N][ldb] pre-transposed
    const float* __restrict__ bias,                   // N (may be null)
    const float* __restrict__ resid,                  // M x N fp32 (EPI==2)
    void* __restrict__ out,                           // M x N
    int M, int N, int K)
{
    __shared__ __align__(16) unsigned short As[2][64 * 64]; // [m][k]
    __shared__ __align__(16) unsigned short Bs[2][64 * 64]; // [n][k]

    const int tid  = threadIdx.x;
    const int lane = tid & 31;
    const int wave = tid >> 5;
    const int wm = wave >> 1, wn = wave & 1;
    const int m0 = blockIdx.y * 64;
    const int n0 = blockIdx.x * 64;

    FragC c[2][2];
#pragma unroll
    for (int i = 0; i < 2; ++i)
#pragma unroll
        for (int j = 0; j < 2; ++j)
#pragma unroll
            for (int e = 0; e < 8; ++e) c[i][j].f[e] = 0.0f;

    const int row = tid >> 1;          // 0..63
    const int cs  = (tid & 1) * 32;    // 0 or 32
    int ma = m0 + row; if (ma >= M) ma = M - 1;
    int nb = n0 + row; if (nb >= N) nb = N - 1;

    const int r   = lane & 15;
    const int kbA = (lane >> 4) * 8;   // A frag: two 8-elem runs
    const int kbB = (lane >> 4) * 16;  // unused: B frag also loads with kbA

    // prologue: stage tile 0 into buffer 0
    async_copy64B(A,  (unsigned)(((size_t)ma * lda + cs) * 2),
                  lds_off(&As[0][row * 64 + cs]));
    async_copy64B(Bt, (unsigned)(((size_t)nb * ldb + cs) * 2),
                  lds_off(&Bs[0][row * 64 + cs]));
    wait_async0();
    __syncthreads();

    int buf = 0;
    for (int k0 = 0; k0 < K; k0 += 64) {
        // ---- prefetch next tile into the alternate buffer (no wait yet)
        if (k0 + 64 < K) {
            async_copy64B(A,  (unsigned)(((size_t)ma * lda + k0 + 64 + cs) * 2),
                          lds_off(&As[buf ^ 1][row * 64 + cs]));
            async_copy64B(Bt, (unsigned)(((size_t)nb * ldb + k0 + 64 + cs) * 2),
                          lds_off(&Bs[buf ^ 1][row * 64 + cs]));
        }
        // ---- compute on current buffer
#pragma unroll
        for (int ks = 0; ks < 64; ks += 32) {
            FragA a[2], b[2];
#pragma unroll
            for (int i = 0; i < 2; ++i) {
                int mr = wm * 32 + i * 16 + r;
                a[i].q[0] = *(const uint4*)(&As[buf][mr * 64 + ks + kbA]);
                a[i].q[1] = *(const uint4*)(&As[buf][mr * 64 + ks + 16 + kbA]);
            }
#pragma unroll
            for (int j = 0; j < 2; ++j) {
                int nr = wn * 32 + j * 16 + r;
                b[j].q[0] = *(const uint4*)(&Bs[buf][nr * 64 + ks + kbA]);
                b[j].q[1] = *(const uint4*)(&Bs[buf][nr * 64 + ks + 16 + kbA]);
            }
#pragma unroll
            for (int i = 0; i < 2; ++i)
#pragma unroll
                for (int j = 0; j < 2; ++j)
                    c[i][j].v = __builtin_amdgcn_wmma_f32_16x16x32_bf16(
                        false, a[i].v, false, b[j].v, (short)0, c[i][j].v, false, false);
        }
        // ---- make prefetched tile visible to all waves, then swap
        wait_async0();
        __syncthreads();
        buf ^= 1;
    }

    // ---- epilogue. C layout: VGPR e -> row e + 8*(lane>=16), col lane&15
    const int rm = (lane >> 4) << 3;
    const int cn = lane & 15;
#pragma unroll
    for (int i = 0; i < 2; ++i) {
#pragma unroll
        for (int j = 0; j < 2; ++j) {
#pragma unroll
            for (int e = 0; e < 8; ++e) {
                int m = m0 + wm * 32 + i * 16 + e + rm;
                int n = n0 + wn * 32 + j * 16 + cn;
                if (m < M && n < N) {
                    float v = c[i][j].f[e];
                    if (bias) v += bias[n];
                    if (EPI == 1) v = 0.5f * v * (1.0f + erff(v * 0.70710678118654752f));
                    if (EPI == 2) v += resid[(size_t)m * N + n];
                    if (OUTBF) ((unsigned short*)out)[(size_t)m * N + n] = f2bf(v);
                    else       ((float*)out)[(size_t)m * N + n] = v;
                }
            }
        }
    }
}

// =====================================================================
// Fused attention: one block per (b,h), 128 threads = 4 waves.
// q,k,v: bf16 [B*S, D], head slice at column h*64.  ao: bf16 [B*S, D].
// scores = (Q K^T)/8 -> softmax (fp32, LDS) -> P V, matmuls via WMMA bf16.
// =====================================================================
__global__ __launch_bounds__(128) void attention_kernel(
    const unsigned short* __restrict__ q, const unsigned short* __restrict__ k,
    const unsigned short* __restrict__ v, unsigned short* __restrict__ ao)
{
    __shared__ __align__(16) float          s_sc[16][224];  // fp32 scores
    __shared__ __align__(16) unsigned short s_p [16][224];  // bf16 softmax P
    __shared__ __align__(16) unsigned short s_vt[64][224];  // bf16 V^T [d][s]

    const int bh = blockIdx.x;
    const int b = bh / H_, h = bh % H_;
    const int tid = threadIdx.x, lane = tid & 31, wave = tid >> 5;
    const float scale = 0.125f;  // 1/sqrt(64)
    const size_t base = (size_t)b * S_ * D_ + (size_t)h * DK_;

    // ---- stage V^T (zero-padded keys 197..223)
    for (int idx = tid; idx < 64 * 224; idx += 128) {
        int d = idx / 224, s = idx % 224;
        s_vt[d][s] = (s < S_) ? v[base + (size_t)s * D_ + d] : (unsigned short)0;
    }
    __syncthreads();

    const int r   = lane & 15;
    const int kbA = (lane >> 4) * 8;
    const int kbB = (lane >> 4) * 16;

    for (int qt = 0; qt < 13; ++qt) {           // 13 query tiles of 16
        // ---- Q fragments (dk=64 -> two k-steps of 32), direct bf16 loads
        FragA aq[2];
        {
            int qr = qt * 16 + r; if (qr >= S_) qr = S_ - 1;
            const unsigned short* qp = q + base + (size_t)qr * D_;
#pragma unroll
            for (int kk = 0; kk < 2; ++kk) {
                aq[kk].q[0] = *(const uint4*)(qp + kk * 32 + kbA);
                aq[kk].q[1] = *(const uint4*)(qp + kk * 32 + 16 + kbA);
            }
        }
        // ---- scores: wave handles key tiles j = wave, wave+4, ...
        for (int j = wave; j < 13; j += 4) {
            FragA bk[2];
            int kr = j * 16 + r; if (kr >= S_) kr = S_ - 1;
            const unsigned short* kp = k + base + (size_t)kr * D_;
#pragma unroll
            for (int kk = 0; kk < 2; ++kk) {
                bk[kk].q[0] = *(const uint4*)(kp + kk * 32 + kbA);
                bk[kk].q[1] = *(const uint4*)(kp + kk * 32 + 16 + kbA);
            }
            FragC c;
#pragma unroll
            for (int e = 0; e < 8; ++e) c.f[e] = 0.0f;
            c.v = __builtin_amdgcn_wmma_f32_16x16x32_bf16(
                false, aq[0].v, false, bk[0].v, (short)0, c.v, false, false);
            c.v = __builtin_amdgcn_wmma_f32_16x16x32_bf16(
                false, aq[1].v, false, bk[1].v, (short)0, c.v, false, false);
#pragma unroll
            for (int e = 0; e < 8; ++e) {
                int m = e + ((lane >> 4) << 3);
                s_sc[m][j * 16 + r] = c.f[e] * scale;
            }
        }
        __syncthreads();

        // ---- softmax: wave handles rows wave*4 .. wave*4+3, lane-parallel cols
        for (int rr = 0; rr < 4; ++rr) {
            int rowi = wave * 4 + rr;
            float mx = -1e30f;
            for (int cc = lane; cc < S_; cc += 32) mx = fmaxf(mx, s_sc[rowi][cc]);
#pragma unroll
            for (int off = 16; off > 0; off >>= 1) mx = fmaxf(mx, __shfl_xor(mx, off, 32));
            float sum = 0.0f;
            for (int cc = lane; cc < S_; cc += 32) {
                float e = __expf(s_sc[rowi][cc] - mx);
                s_sc[rowi][cc] = e;
                sum += e;
            }
#pragma unroll
            for (int off = 16; off > 0; off >>= 1) sum += __shfl_xor(sum, off, 32);
            float inv = 1.0f / sum;
            for (int cc = lane; cc < 224; cc += 32) {
                float pv = (cc < S_) ? s_sc[rowi][cc] * inv : 0.0f;
                s_p[rowi][cc] = f2bf(pv);
            }
        }
        __syncthreads();

        // ---- O = P @ V : wave owns 16 output cols, 7 K-chunks of 32 keys
        {
            int n0 = wave * 16;
            FragC o;
#pragma unroll
            for (int e = 0; e < 8; ++e) o.f[e] = 0.0f;
            for (int kc = 0; kc < 7; ++kc) {
                FragA pa, vb;
                pa.q[0] = *(const uint4*)(&s_p[r][kc * 32 + kbA]);
                pa.q[1] = *(const uint4*)(&s_p[r][kc * 32 + 16 + kbA]);
                vb.q[0] = *(const uint4*)(&s_vt[n0 + r][kc * 32 + kbA]);
                vb.q[1] = *(const uint4*)(&s_vt[n0 + r][kc * 32 + 16 + kbA]);
                o.v = __builtin_amdgcn_wmma_f32_16x16x32_bf16(
                    false, pa.v, false, vb.v, (short)0, o.v, false, false);
            }
#pragma unroll
            for (int e = 0; e < 8; ++e) {
                int m = qt * 16 + e + ((lane >> 4) << 3);
                if (m < S_)
                    ao[(size_t)(b * S_ + m) * D_ + h * DK_ + n0 + (lane & 15)] = f2bf(o.f[e]);
            }
        }
        __syncthreads();
    }
}

// =====================================================================
// Weight convert+transpose: fp32 W[K,N] -> bf16 Wt[N,K]  (once per forward)
// =====================================================================
__global__ __launch_bounds__(256) void transpose_cvt_kernel(
    const float* __restrict__ W, unsigned short* __restrict__ Wt, int K, int N)
{
    __shared__ unsigned short t[32][33];
    const int n0 = blockIdx.x * 32, k0 = blockIdx.y * 32;
    const int tx = threadIdx.x & 31, ty = threadIdx.x >> 5;  // 32 x 8
    for (int i = ty; i < 32; i += 8) {
        int kk = k0 + i, n = n0 + tx;
        float v = (kk < K && n < N) ? W[(size_t)kk * N + n] : 0.0f;
        t[i][tx] = f2bf(v);
    }
    __syncthreads();
    for (int i = ty; i < 32; i += 8) {
        int n = n0 + i, kk = k0 + tx;
        if (n < N && kk < K) Wt[(size_t)n * K + kk] = t[tx][i];
    }
}

// =====================================================================
// LayerNorm over D=768: one block per row, fp32 in -> bf16 out
// =====================================================================
__global__ __launch_bounds__(256) void ln_kernel(
    const float* __restrict__ x, const float* __restrict__ sc,
    const float* __restrict__ bi, unsigned short* __restrict__ out)
{
    __shared__ float red[256];
    const int row = blockIdx.x;
    const float* xr = x + (size_t)row * D_;
    float a = 0.0f;
    for (int i = threadIdx.x; i < D_; i += 256) a += xr[i];
    red[threadIdx.x] = a; __syncthreads();
    for (int st = 128; st > 0; st >>= 1) {
        if (threadIdx.x < st) red[threadIdx.x] += red[threadIdx.x + st];
        __syncthreads();
    }
    float mean = red[0] / (float)D_;
    __syncthreads();
    float vs = 0.0f;
    for (int i = threadIdx.x; i < D_; i += 256) { float d = xr[i] - mean; vs += d * d; }
    red[threadIdx.x] = vs; __syncthreads();
    for (int st = 128; st > 0; st >>= 1) {
        if (threadIdx.x < st) red[threadIdx.x] += red[threadIdx.x + st];
        __syncthreads();
    }
    float inv = rsqrtf(red[0] / (float)D_ + 1e-5f);
    for (int i = threadIdx.x; i < D_; i += 256)
        out[(size_t)row * D_ + i] = f2bf((xr[i] - mean) * inv * sc[i] + bi[i]);
}

// =====================================================================
// Patchify: img[B,3,224,224] -> bf16 [B*196, 768], (py,px,c) fastest order
// =====================================================================
__global__ void im2col_kernel(const float* __restrict__ img,
                              unsigned short* __restrict__ out)
{
    int idx = blockIdx.x * 256 + threadIdx.x;
    const int total = PROWS_ * D_;
    if (idx >= total) return;
    int row = idx / D_, col = idx % D_;
    int b  = row / NPATCH_, gi = row % NPATCH_;
    int gy = gi / GRID_,    gx = gi % GRID_;
    int py = col / (P_ * 3), rem = col % (P_ * 3);
    int px = rem / 3, cch = rem % 3;
    out[idx] = f2bf(img[(((size_t)b * 3 + cch) * IMG_ + gy * P_ + py) * IMG_ + gx * P_ + px]);
}

// x[b, s, :] = (s==0 ? cls : patch_gemm[b, s-1, :]) + pos_emb[s, :]
__global__ void assemble_kernel(const float* __restrict__ ptmp,
                                const float* __restrict__ cls,
                                const float* __restrict__ pos,
                                float* __restrict__ x)
{
    int idx = blockIdx.x * 256 + threadIdx.x;
    const int total = ROWS_ * D_;
    if (idx >= total) return;
    int row = idx / D_, d = idx % D_;
    int b = row / S_, s = row % S_;
    float v0 = (s == 0) ? cls[d] : ptmp[((size_t)b * NPATCH_ + (s - 1)) * D_ + d];
    x[idx] = v0 + pos[(size_t)s * D_ + d];
}

// =====================================================================
extern "C" void kernel_launch(void* const* d_in, const int* in_sizes, int n_in,
                              void* d_out, int out_size, void* d_ws, size_t ws_size,
                              hipStream_t stream)
{
    (void)in_sizes; (void)n_in; (void)out_size; (void)ws_size;
    const float* img    = (const float*)d_in[0];
    const float* proj_w = (const float*)d_in[1];
    const float* proj_b = (const float*)d_in[2];
    const float* cls    = (const float*)d_in[3];
    const float* pos    = (const float*)d_in[4];
    const float* ln1_s  = (const float*)d_in[5];
    const float* ln1_b  = (const float*)d_in[6];
    const float* wq = (const float*)d_in[7];  const float* bq = (const float*)d_in[8];
    const float* wk = (const float*)d_in[9];  const float* bk = (const float*)d_in[10];
    const float* wv = (const float*)d_in[11]; const float* bv = (const float*)d_in[12];
    const float* wo = (const float*)d_in[13]; const float* bo = (const float*)d_in[14];
    const float* ln2_s = (const float*)d_in[15]; const float* ln2_b = (const float*)d_in[16];
    const float* w1 = (const float*)d_in[17]; const float* b1 = (const float*)d_in[18];
    const float* w2 = (const float*)d_in[19]; const float* b2 = (const float*)d_in[20];
    const float* fns = (const float*)d_in[21]; const float* fnb = (const float*)d_in[22];
    const float* hw  = (const float*)d_in[23]; const float* hb  = (const float*)d_in[24];
    float* outp = (float*)d_out;

    // ---- workspace carve-up (256B aligned)
    char* ws = (char*)d_ws;
    auto alloc = [&](size_t bytes) {
        char* p = ws; ws += (bytes + 255) & ~(size_t)255; return p;
    };
    // activations
    float*          x    = (float*)         alloc((size_t)ROWS_  * D_ * 4);
    unsigned short* xn   = (unsigned short*)alloc((size_t)ROWS_  * D_ * 2);
    unsigned short* qb   = (unsigned short*)alloc((size_t)ROWS_  * D_ * 2);
    unsigned short* kb   = (unsigned short*)alloc((size_t)ROWS_  * D_ * 2);
    unsigned short* vb   = (unsigned short*)alloc((size_t)ROWS_  * D_ * 2);
    unsigned short* aob  = (unsigned short*)alloc((size_t)ROWS_  * D_ * 2);
    unsigned short* hbuf = (unsigned short*)alloc((size_t)ROWS_  * F_ * 2);
    unsigned short* pimg = (unsigned short*)alloc((size_t)PROWS_ * D_ * 2);
    float*          ptmp = (float*)         alloc((size_t)PROWS_ * D_ * 4);
    // pre-transposed bf16 weights [N][K]
    unsigned short* wt_proj = (unsigned short*)alloc((size_t)D_ * D_ * 2);
    unsigned short* wt_q    = (unsigned short*)alloc((size_t)L_ * D_ * D_ * 2);
    unsigned short* wt_k    = (unsigned short*)alloc((size_t)L_ * D_ * D_ * 2);
    unsigned short* wt_v    = (unsigned short*)alloc((size_t)L_ * D_ * D_ * 2);
    unsigned short* wt_o    = (unsigned short*)alloc((size_t)L_ * D_ * D_ * 2);
    unsigned short* wt_1    = (unsigned short*)alloc((size_t)L_ * D_ * F_ * 2);
    unsigned short* wt_2    = (unsigned short*)alloc((size_t)L_ * F_ * D_ * 2);
    unsigned short* wt_h    = (unsigned short*)alloc((size_t)NC_ * D_ * 2);

    auto T = [&](const float* W, unsigned short* Wt, int K, int N) {
        dim3 g((N + 31) / 32, (K + 31) / 32);
        transpose_cvt_kernel<<<g, 256, 0, stream>>>(W, Wt, K, N);
    };

    // ---- weight convert + transpose (once per forward)
    T(proj_w, wt_proj, D_, D_);
    for (int l = 0; l < L_; ++l) {
        T(wq + (size_t)l * D_ * D_, wt_q + (size_t)l * D_ * D_, D_, D_);
        T(wk + (size_t)l * D_ * D_, wt_k + (size_t)l * D_ * D_, D_, D_);
        T(wv + (size_t)l * D_ * D_, wt_v + (size_t)l * D_ * D_, D_, D_);
        T(wo + (size_t)l * D_ * D_, wt_o + (size_t)l * D_ * D_, D_, D_);
        T(w1 + (size_t)l * D_ * F_, wt_1 + (size_t)l * D_ * F_, D_, F_);
        T(w2 + (size_t)l * F_ * D_, wt_2 + (size_t)l * F_ * D_, F_, D_);
    }
    T(hw, wt_h, D_, NC_);

    const dim3 gD((D_  + 63) / 64, (ROWS_  + 63) / 64);
    const dim3 gF((F_  + 63) / 64, (ROWS_  + 63) / 64);
    const dim3 gP((D_  + 63) / 64, (PROWS_ + 63) / 64);
    const dim3 gH((NC_ + 63) / 64, (B_     + 63) / 64);

    // ---- patch embedding
    im2col_kernel<<<(PROWS_ * D_ + 255) / 256, 256, 0, stream>>>(img, pimg);
    gemm_bf16_kernel<0, 0><<<gP, 128, 0, stream>>>(
        pimg, D_, wt_proj, D_, proj_b, nullptr, ptmp, PROWS_, D_, D_);
    assemble_kernel<<<(ROWS_ * D_ + 255) / 256, 256, 0, stream>>>(ptmp, cls, pos, x);

    // ---- transformer layers
    for (int l = 0; l < L_; ++l) {
        ln_kernel<<<ROWS_, 256, 0, stream>>>(x, ln1_s + l * D_, ln1_b + l * D_, xn);
        gemm_bf16_kernel<0, 1><<<gD, 128, 0, stream>>>(
            xn, D_, wt_q + (size_t)l * D_ * D_, D_, bq + l * D_, nullptr, qb, ROWS_, D_, D_);
        gemm_bf16_kernel<0, 1><<<gD, 128, 0, stream>>>(
            xn, D_, wt_k + (size_t)l * D_ * D_, D_, bk + l * D_, nullptr, kb, ROWS_, D_, D_);
        gemm_bf16_kernel<0, 1><<<gD, 128, 0, stream>>>(
            xn, D_, wt_v + (size_t)l * D_ * D_, D_, bv + l * D_, nullptr, vb, ROWS_, D_, D_);
        attention_kernel<<<B_ * H_, 128, 0, stream>>>(qb, kb, vb, aob);
        gemm_bf16_kernel<2, 0><<<gD, 128, 0, stream>>>(
            aob, D_, wt_o + (size_t)l * D_ * D_, D_, bo + l * D_, x, x, ROWS_, D_, D_);
        ln_kernel<<<ROWS_, 256, 0, stream>>>(x, ln2_s + l * D_, ln2_b + l * D_, xn);
        gemm_bf16_kernel<1, 1><<<gF, 128, 0, stream>>>(
            xn, D_, wt_1 + (size_t)l * D_ * F_, D_, b1 + l * F_, nullptr, hbuf, ROWS_, F_, D_);
        gemm_bf16_kernel<2, 0><<<gD, 128, 0, stream>>>(
            hbuf, F_, wt_2 + (size_t)l * F_ * D_, F_, b2 + l * D_, x, x, ROWS_, D_, F_);
    }

    // ---- final LN + classifier head (cls rows gathered via lda = S*D)
    ln_kernel<<<ROWS_, 256, 0, stream>>>(x, fns, fnb, xn);
    gemm_bf16_kernel<0, 0><<<gH, 128, 0, stream>>>(
        xn, S_ * D_, wt_h, D_, hb, nullptr, outp, B_, NC_, D_);
}
